// QLinearLPBQ_60773787239146
// MI455X (gfx1250) — matrix operand order repaired and checked by reading, not compiled, not run on hardware
//
#include <hip/hip_runtime.h>

// QLinear with low-precision block quantization (LPBQ), fused on-the-fly
// dequant + f16 WMMA GEMM for MI455X (gfx1250, wave32).
//
// out[128,11008] = x[128,4096] @ dequant(W[11008,4096])^T + bias
//
// HBM-bound on the 180MB f32 weight stream (~8us at 23.3 TB/s). Weight is
// read once per workgroup tile in pass 1 (scales) and re-read from L2
// (192MB global L2 holds nearly the whole weight) in pass 2 (dequant+GEMM).

typedef __attribute__((ext_vector_type(16))) _Float16 v16h;
typedef __attribute__((ext_vector_type(8)))  _Float16 v8h;
typedef __attribute__((ext_vector_type(8)))  float    v8f;
typedef __attribute__((ext_vector_type(4)))  float    f32x4;

#define IN_CH   4096
#define OUT_CH  11008
#define M_ROWS  128
#define QBLK    64                 // quantization block along K
#define NBLK    (IN_CH / QBLK)     // 64 blocks per weight row
#define NTILE   64                 // output channels per workgroup
#define KTILE   32                 // WMMA K per step
#define THREADS 256                // 8 wave32 waves

static __device__ __forceinline__ v16h cat8(v8h lo, v8h hi) {
  return __builtin_shufflevector(lo, hi, 0,1,2,3,4,5,6,7,8,9,10,11,12,13,14,15);
}

__global__ __launch_bounds__(THREADS)
void qlinear_lpbq_wmma(const float* __restrict__ x,
                       const float* __restrict__ w,
                       const float* __restrict__ bias,
                       float* __restrict__ out)
{
  // sc: per-(row,block) absmax in pass 1, overwritten in-place with s1_recon.
  __shared__ __align__(16) float     sc[NTILE * NBLK];   // 16 KB
  __shared__             float     s2row[NTILE];         // 256 B
  __shared__ __align__(16) _Float16 As[M_ROWS * KTILE];  // 8 KB  (x tile, f16)
  __shared__ __align__(16) _Float16 Bs[NTILE * KTILE];   // 4 KB  (deq W tile, f16)

  const int tid  = threadIdx.x;
  const int n0   = blockIdx.x * NTILE;      // first output channel of tile
  const int lane = tid & 31;
  const int wave = tid >> 5;
  const int wm   = (wave & 3) * 32;         // wave M base: 0/32/64/96
  const int wn   = (wave >> 2) * 32;        // wave N base: 0/32

  // ---------------- Pass 1: per-(row,block) absmax ----------------
  for (int j = tid; j < NTILE * NBLK; j += THREADS) {
    const int row = j >> 6;                 // NBLK == 64
    const int blk = j & (NBLK - 1);
    const float* p = w + (size_t)(n0 + row) * IN_CH + blk * QBLK;
    float m = 0.f;
    #pragma unroll
    for (int i = 0; i < QBLK / 4; ++i) {
      f32x4 v = *(const f32x4*)(p + i * 4);
      m = fmaxf(m, fmaxf(fmaxf(fabsf(v.x), fabsf(v.y)),
                         fmaxf(fabsf(v.z), fabsf(v.w))));
    }
    sc[j] = m;
  }
  __syncthreads();

  // per-row second-level scale: s2 = clip(max_blk(s1)/15, 1e-8)
  if (tid < NTILE) {
    float m = 0.f;
    #pragma unroll 8
    for (int b = 0; b < NBLK; ++b) m = fmaxf(m, sc[tid * NBLK + b]);
    const float s1max = fmaxf(m * (1.f / 7.f), 1e-8f);
    s2row[tid] = fmaxf(s1max * (1.f / 15.f), 1e-8f);
  }
  __syncthreads();

  // s1_recon = clip(round(s1/s2), 0, 15) * s2   (in-place over absmax)
  for (int j = tid; j < NTILE * NBLK; j += THREADS) {
    const int   row = j >> 6;
    const float s1  = fmaxf(sc[j] * (1.f / 7.f), 1e-8f);
    const float s2  = s2row[row];
    const float q   = fminf(fmaxf(rintf(s1 / s2), 0.f), 15.f);
    sc[j] = q * s2;
  }
  __syncthreads();

  // ---------------- Pass 2: K loop, dequant -> LDS -> WMMA ----------------
  v8f acc[2][2] = {};

  for (int k0 = 0; k0 < IN_CH; k0 += KTILE) {
    // Stage A tile: x[0..127][k0..k0+32) f32 -> f16. 16 halves per thread.
    {
      const int idx = tid * 16;             // 4096 elements total
      const int row = idx >> 5;
      const int col = idx & 31;
      const float* p = x + (size_t)row * IN_CH + k0 + col;
      _Float16* q = &As[row * KTILE + col];
      #pragma unroll
      for (int i = 0; i < 4; ++i) {
        f32x4 v = *(const f32x4*)(p + i * 4);
        q[i * 4 + 0] = (_Float16)v.x;
        q[i * 4 + 1] = (_Float16)v.y;
        q[i * 4 + 2] = (_Float16)v.z;
        q[i * 4 + 3] = (_Float16)v.w;
      }
    }
    // Stage B tile: dequant W rows [n0, n0+64) cols [k0, k0+32) -> f16.
    // KTILE (32) divides QBLK (64) and k0 is 32-aligned, so one s1_recon
    // per row covers the whole k-step: blk = k0 >> 6.
    {
      const int idx = tid * 8;              // 2048 elements total
      const int row = idx >> 5;
      const int col = idx & 31;
      const float s1r = sc[row * NBLK + (k0 >> 6)];
      const float inv = (s1r > 0.f) ? (1.f / s1r) : 0.f;
      const float* p = w + (size_t)(n0 + row) * IN_CH + k0 + col;
      _Float16* q = &Bs[row * KTILE + col];
      #pragma unroll
      for (int i = 0; i < 2; ++i) {
        f32x4 v = *(const f32x4*)(p + i * 4);
        q[i * 4 + 0] = (_Float16)(fminf(fmaxf(rintf(v.x * inv), -8.f), 7.f) * s1r);
        q[i * 4 + 1] = (_Float16)(fminf(fmaxf(rintf(v.y * inv), -8.f), 7.f) * s1r);
        q[i * 4 + 2] = (_Float16)(fminf(fmaxf(rintf(v.z * inv), -8.f), 7.f) * s1r);
        q[i * 4 + 3] = (_Float16)(fminf(fmaxf(rintf(v.w * inv), -8.f), 7.f) * s1r);
      }
    }
    __syncthreads();

    // Fragment loads per CDNA5 ISA VGPR layouts (CDNA5 ISA §7.12.2):
    // A (16x32 f16): lane L holds row L%16; K halves [(L/16)*8, +8) and +16.
    // B (32x16 f16): lane L holds col L%16; K halves [(L/16)*8, +8) and +16 (same as A).
    const int kh = (lane >> 4) * 8;                 // A K-chunk base
    const int r16 = lane & 15;

    const int am0 = (wm + r16) * KTILE;
    const int am1 = (wm + 16 + r16) * KTILE;
    v16h a0 = cat8(*(const v8h*)&As[am0 + kh], *(const v8h*)&As[am0 + kh + 16]);
    v16h a1 = cat8(*(const v8h*)&As[am1 + kh], *(const v8h*)&As[am1 + kh + 16]);


    v16h b0 = cat8(*(const v8h*)&Bs[(wn +      r16) * KTILE + kh], *(const v8h*)&Bs[(wn +      r16) * KTILE + kh + 16]);
    v16h b1 = cat8(*(const v8h*)&Bs[(wn + 16 + r16) * KTILE + kh], *(const v8h*)&Bs[(wn + 16 + r16) * KTILE + kh + 16]);

    acc[0][0] = __builtin_amdgcn_wmma_f32_16x16x32_f16(false, a0, false, b0,
                                                       (short)0, acc[0][0], false, false);
    acc[0][1] = __builtin_amdgcn_wmma_f32_16x16x32_f16(false, a0, false, b1,
                                                       (short)0, acc[0][1], false, false);
    acc[1][0] = __builtin_amdgcn_wmma_f32_16x16x32_f16(false, a1, false, b0,
                                                       (short)0, acc[1][0], false, false);
    acc[1][1] = __builtin_amdgcn_wmma_f32_16x16x32_f16(false, a1, false, b1,
                                                       (short)0, acc[1][1], false, false);

    __syncthreads();
  }

  // ---------------- Epilogue: bias + store ----------------
  // C layout: VGPR r -> M=r (lanes 0-15) / M=r+8 (lanes 16-31); N = lane%16.
  const int r16  = lane & 15;
  const int mofs = (lane >> 4) << 3;
  #pragma unroll
  for (int ni = 0; ni < 2; ++ni) {
    const int n  = n0 + wn + ni * 16 + r16;
    const float bv = bias[n];
    #pragma unroll
    for (int mi = 0; mi < 2; ++mi) {
      const int mbase = wm + mi * 16 + mofs;
      #pragma unroll
      for (int r = 0; r < 8; ++r) {
        out[(size_t)(mbase + r) * OUT_CH + n] = acc[mi][ni][r] + bv;
      }
    }
  }
}

extern "C" void kernel_launch(void* const* d_in, const int* in_sizes, int n_in,
                              void* d_out, int out_size, void* d_ws, size_t ws_size,
                              hipStream_t stream) {
  const float* x    = (const float*)d_in[0];   // [1,128,4096]
  const float* wgt  = (const float*)d_in[1];   // [11008,4096]
  const float* bias = (const float*)d_in[2];   // [11008]
  float* out = (float*)d_out;                  // [1,128,11008]

  dim3 grid(OUT_CH / NTILE);                   // 172 workgroups
  dim3 block(THREADS);                         // 8 wave32 waves
  qlinear_lpbq_wmma<<<grid, block, 0, stream>>>(x, wgt, bias, out);
}
